// S_CLSTM_DANN_15324443312128
// MI455X (gfx1250) — matrix operand order repaired and checked by reading, not compiled, not run on hardware
//
#include <hip/hip_runtime.h>

// ---------------------------------------------------------------------------
// CDNA5 (gfx1250, wave32) implementation of the S-CLSTM-DANN forward pass.
// Heavy recurrence (2 SLSTM layers x 400 steps) runs on v_wmma_f32_16x16x32_bf16.
// ---------------------------------------------------------------------------

typedef __attribute__((ext_vector_type(16))) __bf16 bf16x16;
typedef __attribute__((ext_vector_type(8)))  __bf16 bf16x8;
typedef __attribute__((ext_vector_type(8)))  float  f32x8;

#define TT   400
#define BB   128
#define CCH  14
#define OC   64
#define HH   512
#define G4   2048
#define BETA 0.9f
#define BN_EPS 1e-5f

__device__ __forceinline__ float sigm(float x) { return 1.0f / (1.0f + __expf(-x)); }

// load 16 consecutive-ish bf16 (two 16B chunks) into a v16 for WMMA operands
__device__ __forceinline__ bf16x16 ld_v16(const __bf16* p0, const __bf16* p1) {
  bf16x8 a = *(const bf16x8*)p0;
  bf16x8 b = *(const bf16x8*)p1;
  return __builtin_shufflevector(a, b, 0,1,2,3,4,5,6,7,8,9,10,11,12,13,14,15);
}

// ---------------------------------------------------------------------------
// conv1d over time (pad 2, k=5): y[b][oc][t]
// ---------------------------------------------------------------------------
__global__ __launch_bounds__(64)
void conv_kernel(const float* __restrict__ x, const float* __restrict__ cw,
                 float* __restrict__ y) {
  const int t = blockIdx.x, b = blockIdx.y, oc = threadIdx.x;
  __shared__ float xs[5 * CCH];
  for (int i = threadIdx.x; i < 5 * CCH; i += 64) {
    int k = i / CCH, ic = i % CCH;
    int tt = t + k - 2;
    xs[i] = (tt >= 0 && tt < TT) ? x[(size_t)(tt * BB + b) * CCH + ic] : 0.0f;
  }
  __syncthreads();
  float acc = 0.0f;
  #pragma unroll
  for (int k = 0; k < 5; ++k)
    for (int ic = 0; ic < CCH; ++ic)
      acc += xs[k * CCH + ic] * cw[((oc * CCH) + ic) * 5 + k];
  y[((size_t)(b * OC + oc)) * TT + t] = acc;
}

// ---------------------------------------------------------------------------
// BatchNorm stats (training mode, biased var): per-channel scale/shift
// ---------------------------------------------------------------------------
__global__ __launch_bounds__(256)
void bnstats_kernel(const float* __restrict__ y, const float* __restrict__ g,
                    const float* __restrict__ b, float* __restrict__ scale,
                    float* __restrict__ shift) {
  const int oc = blockIdx.x;
  __shared__ float ssum[256], ssq[256];
  float s = 0.0f, q = 0.0f;
  const int N = BB * TT;
  for (int i = threadIdx.x; i < N; i += 256) {
    int bb = i / TT, t = i % TT;
    float v = y[((size_t)(bb * OC + oc)) * TT + t];
    s += v; q += v * v;
  }
  ssum[threadIdx.x] = s; ssq[threadIdx.x] = q;
  __syncthreads();
  for (int st = 128; st > 0; st >>= 1) {
    if (threadIdx.x < st) { ssum[threadIdx.x] += ssum[threadIdx.x + st];
                            ssq[threadIdx.x]  += ssq[threadIdx.x + st]; }
    __syncthreads();
  }
  if (threadIdx.x == 0) {
    float mu  = ssum[0] / N;
    float var = ssq[0] / N - mu * mu;
    float sc  = g[oc] * rsqrtf(var + BN_EPS);
    scale[oc] = sc;
    shift[oc] = b[oc] - mu * sc;
  }
}

// ---------------------------------------------------------------------------
// LIF1 recurrence (pointwise over time; independent of SLSTM state):
// precompute all spike inputs spk_c[t][b][oc] as bf16
// ---------------------------------------------------------------------------
__global__ __launch_bounds__(256)
void lif_kernel(const float* __restrict__ y, const float* __restrict__ scale,
                const float* __restrict__ shift, const float* __restrict__ thr_p,
                __bf16* __restrict__ spkc) {
  const int id = blockIdx.x * 256 + threadIdx.x;   // 8192 = 128*64
  const int b = id / OC, oc = id % OC;
  const float sc = scale[oc], sh = shift[oc], thr = *thr_p;
  const float* yr = y + ((size_t)(b * OC + oc)) * TT;
  float memv = 0.0f;
  for (int t = 0; t < TT; ++t) {
    float xt = yr[t] * sc + sh;
    float reset = (memv - thr) > 0.0f ? 1.0f : 0.0f;
    memv = BETA * memv + xt - reset * thr;
    spkc[(size_t)t * (BB * OC) + b * OC + oc] =
        (__bf16)((memv - thr) > 0.0f ? 1.0f : 0.0f);
  }
}

// ---------------------------------------------------------------------------
// utility: f32 -> bf16 weight conversion, bias fuse
// ---------------------------------------------------------------------------
__global__ void cvt_bf16(const float* __restrict__ s, __bf16* __restrict__ d, int n) {
  int i = blockIdx.x * 256 + threadIdx.x;
  if (i < n) d[i] = (__bf16)s[i];
}
__global__ void bias_fuse(const float* __restrict__ a, const float* __restrict__ b,
                          float* __restrict__ d, int n) {
  int i = blockIdx.x * 256 + threadIdx.x;
  if (i < n) d[i] = a[i] + b[i];
}

// ---------------------------------------------------------------------------
// One SLSTM step: gates = Xin @ Wih^T + Hbf @ Whh^T + bias, fused state update.
// Grid (16, 4): block owns (m0..m0+31) x (h0..h0+31) for ALL 4 gates.
// 8 waves: wave w -> gate (w>>1), n-subtile (w&1); each wave = two 16x16 accs.
// WMMA: v_wmma_f32_16x16x32_bf16, K-loop over hidden(512) + input(KIN).
// ---------------------------------------------------------------------------
template <int KIN>
__global__ __launch_bounds__(256)
void slstm_step(const __bf16* __restrict__ Xin,   // [128, KIN]
                const __bf16* __restrict__ Hbf,   // [128, 512] prev mem (bf16)
                const __bf16* __restrict__ Wih,   // [2048, KIN]
                const __bf16* __restrict__ Whh,   // [2048, 512]
                const float*  __restrict__ bias,  // [2048]
                float* __restrict__ syn, float* __restrict__ mem,
                __bf16* __restrict__ hbf_out,     // [128,512] mem bf16 (next t)
                __bf16* __restrict__ spk_out,     // [128,512] spike (next layer)
                float* __restrict__ feat_acc,     // nullptr or [128,512]
                const float* __restrict__ thr_p) {
  __shared__ float gl[4][32][32];                 // 16 KB gate exchange

  const int tid   = threadIdx.x;
  const int lane  = tid & 31;
  const int wave  = tid >> 5;
  const int gate  = wave >> 1;
  const int ntile = wave & 1;
  const int h0    = blockIdx.x * 32;
  const int m0    = blockIdx.y * 32;

  const int r  = lane & 15;   // A row within 16
  const int hi = lane >> 4;   // half-wave selector
  const int nl = lane & 15;   // B/D column within 16
  const int kb = hi * 16;     // unused; B K-bases are 8*hi and 16 + 8*hi
  const int ng = gate * HH + h0 + ntile * 16 + nl;  // global gate column

  f32x8 acc0 = {}; f32x8 acc1 = {};

  // ---- hidden GEMM: K = 512 ----
  const __bf16* wr = Whh + (size_t)ng * HH;
  const __bf16* a0 = Hbf + (size_t)(m0 + r) * HH;
  const __bf16* a1 = Hbf + (size_t)(m0 + 16 + r) * HH;
  #pragma unroll 4
  for (int k0 = 0; k0 < HH; k0 += 32) {
    bf16x16 bm  = ld_v16(wr + k0 + hi * 8,      wr + k0 + 16 + hi * 8);
    bf16x16 am0 = ld_v16(a0 + k0 + hi * 8,      a0 + k0 + 16 + hi * 8);
    bf16x16 am1 = ld_v16(a1 + k0 + hi * 8,      a1 + k0 + 16 + hi * 8);
    acc0 = __builtin_amdgcn_wmma_f32_16x16x32_bf16(false, am0, false, bm,
                                                   (short)0, acc0, false, false);
    acc1 = __builtin_amdgcn_wmma_f32_16x16x32_bf16(false, am1, false, bm,
                                                   (short)0, acc1, false, false);
  }
  // ---- input GEMM: K = KIN ----
  const __bf16* wi = Wih + (size_t)ng * KIN;
  const __bf16* x0 = Xin + (size_t)(m0 + r) * KIN;
  const __bf16* x1 = Xin + (size_t)(m0 + 16 + r) * KIN;
  #pragma unroll 2
  for (int k0 = 0; k0 < KIN; k0 += 32) {
    bf16x16 bm  = ld_v16(wi + k0 + hi * 8,      wi + k0 + 16 + hi * 8);
    bf16x16 am0 = ld_v16(x0 + k0 + hi * 8,      x0 + k0 + 16 + hi * 8);
    bf16x16 am1 = ld_v16(x1 + k0 + hi * 8,      x1 + k0 + 16 + hi * 8);
    acc0 = __builtin_amdgcn_wmma_f32_16x16x32_bf16(false, am0, false, bm,
                                                   (short)0, acc0, false, false);
    acc1 = __builtin_amdgcn_wmma_f32_16x16x32_bf16(false, am1, false, bm,
                                                   (short)0, acc1, false, false);
  }

  // ---- exchange gates through LDS (C/D layout: row = q + 8*hi, col = nl) ----
  const int nloc = ntile * 16 + nl;
  #pragma unroll
  for (int q = 0; q < 8; ++q) {
    gl[gate][8 * hi + q][nloc]      = acc0[q];
    gl[gate][16 + 8 * hi + q][nloc] = acc1[q];
  }
  __syncthreads();

  // ---- fused SLSTM pointwise update (4 elements / thread) ----
  const float thr = *thr_p;
  #pragma unroll
  for (int q = 0; q < 4; ++q) {
    int idx = tid + q * 256;
    int ml = idx >> 5, nc = idx & 31;
    int h = h0 + nc;
    size_t off = (size_t)(m0 + ml) * HH + h;
    float gi = gl[0][ml][nc] + bias[h];
    float gf = gl[1][ml][nc] + bias[HH + h];
    float gg = gl[2][ml][nc] + bias[2 * HH + h];
    float go = gl[3][ml][nc] + bias[3 * HH + h];
    float ii = sigm(gi), ff = sigm(gf), cg = tanhf(gg), oo = sigm(go);
    float sn = ff * syn[off] + ii * cg;
    float mp = mem[off];
    float reset = (mp - thr) > 0.0f ? 1.0f : 0.0f;
    float mn = oo * tanhf(sn) - reset * thr;
    syn[off] = sn;
    mem[off] = mn;
    hbf_out[off] = (__bf16)mn;
    spk_out[off] = (__bf16)((mn - thr) > 0.0f ? 1.0f : 0.0f);
    if (feat_acc) feat_acc[off] += mn;
  }
}

// ---------------------------------------------------------------------------
// Head: features = feat_acc/T; gesture = feat @ wg^T + bg;
// domain = H(feat @ wd1^T + bd1 - thr) @ wd2^T + bd2
// ---------------------------------------------------------------------------
__global__ __launch_bounds__(64)
void head_kernel(const float* __restrict__ feat_acc,
                 const float* __restrict__ wg,  const float* __restrict__ bg,
                 const float* __restrict__ wd1, const float* __restrict__ bd1,
                 const float* __restrict__ wd2, const float* __restrict__ bd2,
                 const float* __restrict__ thr_p, float* __restrict__ out) {
  const int b = blockIdx.x, tid = threadIdx.x;
  __shared__ float spkd[64];
  const float invT = 1.0f / (float)TT;
  const float* f = feat_acc + (size_t)b * HH;
  // domain hidden (64 units, one per thread)
  {
    const float* w = wd1 + (size_t)tid * HH;
    float acc = 0.0f;
    for (int k = 0; k < HH; ++k) acc += f[k] * w[k];
    float dh = acc * invT + bd1[tid];
    spkd[tid] = (dh - *thr_p) > 0.0f ? 1.0f : 0.0f;
  }
  // gesture (8 classes)
  if (tid < 8) {
    const float* w = wg + (size_t)tid * HH;
    float acc = 0.0f;
    for (int k = 0; k < HH; ++k) acc += f[k] * w[k];
    out[b * 8 + tid] = acc * invT + bg[tid];
  }
  __syncthreads();
  // domain (10 classes)
  if (tid < 10) {
    float acc = 0.0f;
    #pragma unroll
    for (int j = 0; j < 64; ++j) acc += spkd[j] * wd2[tid * 64 + j];
    out[BB * 8 + b * 10 + tid] = acc + bd2[tid];
  }
}

// ---------------------------------------------------------------------------
// workspace layout (bytes, all 256-aligned)
// ---------------------------------------------------------------------------
static constexpr size_t OFF_Y     = 0;                                    // 13107200
static constexpr size_t OFF_SCALE = OFF_Y     + (size_t)BB * OC * TT * 4;
static constexpr size_t OFF_SHIFT = OFF_SCALE + 256;
static constexpr size_t OFF_SPKC  = OFF_SHIFT + 256;                      // 6553600
static constexpr size_t OFF_WIH1  = OFF_SPKC  + (size_t)TT * BB * OC * 2;
static constexpr size_t OFF_WHH1  = OFF_WIH1  + (size_t)G4 * OC * 2;
static constexpr size_t OFF_WIH2  = OFF_WHH1  + (size_t)G4 * HH * 2;
static constexpr size_t OFF_WHH2  = OFF_WIH2  + (size_t)G4 * HH * 2;
static constexpr size_t OFF_B1    = OFF_WHH2  + (size_t)G4 * HH * 2;
static constexpr size_t OFF_B2    = OFF_B1    + (size_t)G4 * 4;
static constexpr size_t OFF_STATE = OFF_B2    + (size_t)G4 * 4;
static constexpr size_t SB_F32    = (size_t)BB * HH * 4;                  // 262144
static constexpr size_t SB_BF16   = (size_t)BB * HH * 2;                  // 131072
static constexpr size_t OFF_SYN1  = OFF_STATE;
static constexpr size_t OFF_MEM1  = OFF_SYN1 + SB_F32;
static constexpr size_t OFF_SYN2  = OFF_MEM1 + SB_F32;
static constexpr size_t OFF_MEM2  = OFF_SYN2 + SB_F32;
static constexpr size_t OFF_FEAT  = OFF_MEM2 + SB_F32;
static constexpr size_t OFF_M1BF0 = OFF_FEAT + SB_F32;
static constexpr size_t OFF_M1BF1 = OFF_M1BF0 + SB_BF16;
static constexpr size_t OFF_M2BF0 = OFF_M1BF1 + SB_BF16;
static constexpr size_t OFF_M2BF1 = OFF_M2BF0 + SB_BF16;
static constexpr size_t OFF_SPK1  = OFF_M2BF1 + SB_BF16;
static constexpr size_t OFF_SPK2  = OFF_SPK1 + SB_BF16;
static constexpr size_t STATE_BYTES = OFF_SPK2 + SB_BF16 - OFF_STATE;

extern "C" void kernel_launch(void* const* d_in, const int* in_sizes, int n_in,
                              void* d_out, int out_size, void* d_ws, size_t ws_size,
                              hipStream_t stream) {
  const float* x      = (const float*)d_in[0];
  const float* conv_w = (const float*)d_in[1];
  const float* bn_g   = (const float*)d_in[2];
  const float* bn_b   = (const float*)d_in[3];
  const float* w_ih1  = (const float*)d_in[4];
  const float* w_hh1  = (const float*)d_in[5];
  const float* b_ih1  = (const float*)d_in[6];
  const float* b_hh1  = (const float*)d_in[7];
  const float* w_ih2  = (const float*)d_in[8];
  const float* w_hh2  = (const float*)d_in[9];
  const float* b_ih2  = (const float*)d_in[10];
  const float* b_hh2  = (const float*)d_in[11];
  const float* wg     = (const float*)d_in[12];
  const float* bg     = (const float*)d_in[13];
  const float* wd1    = (const float*)d_in[14];
  const float* bd1    = (const float*)d_in[15];
  const float* wd2    = (const float*)d_in[16];
  const float* bd2    = (const float*)d_in[17];
  const float* thr_l  = (const float*)d_in[18];
  const float* thr1   = (const float*)d_in[19];
  const float* thr2   = (const float*)d_in[20];
  const float* thrd   = (const float*)d_in[21];

  char* ws = (char*)d_ws;
  float*  Y     = (float*)(ws + OFF_Y);
  float*  SCALE = (float*)(ws + OFF_SCALE);
  float*  SHIFT = (float*)(ws + OFF_SHIFT);
  __bf16* SPKC  = (__bf16*)(ws + OFF_SPKC);
  __bf16* WIH1  = (__bf16*)(ws + OFF_WIH1);
  __bf16* WHH1  = (__bf16*)(ws + OFF_WHH1);
  __bf16* WIH2  = (__bf16*)(ws + OFF_WIH2);
  __bf16* WHH2  = (__bf16*)(ws + OFF_WHH2);
  float*  B1    = (float*)(ws + OFF_B1);
  float*  B2    = (float*)(ws + OFF_B2);
  float*  SYN1  = (float*)(ws + OFF_SYN1);
  float*  MEM1  = (float*)(ws + OFF_MEM1);
  float*  SYN2  = (float*)(ws + OFF_SYN2);
  float*  MEM2  = (float*)(ws + OFF_MEM2);
  float*  FEAT  = (float*)(ws + OFF_FEAT);
  __bf16* M1BF[2] = { (__bf16*)(ws + OFF_M1BF0), (__bf16*)(ws + OFF_M1BF1) };
  __bf16* M2BF[2] = { (__bf16*)(ws + OFF_M2BF0), (__bf16*)(ws + OFF_M2BF1) };
  __bf16* SPK1  = (__bf16*)(ws + OFF_SPK1);
  __bf16* SPK2  = (__bf16*)(ws + OFF_SPK2);

  // --- frontend ---
  conv_kernel<<<dim3(TT, BB), 64, 0, stream>>>(x, conv_w, Y);
  bnstats_kernel<<<OC, 256, 0, stream>>>(Y, bn_g, bn_b, SCALE, SHIFT);
  lif_kernel<<<(BB * OC) / 256, 256, 0, stream>>>(Y, SCALE, SHIFT, thr_l, SPKC);

  // --- weight prep (f32 -> bf16, keep native [4H, K] layout = WMMA B layout) ---
  cvt_bf16<<<(G4 * OC + 255) / 256, 256, 0, stream>>>(w_ih1, WIH1, G4 * OC);
  cvt_bf16<<<(G4 * HH + 255) / 256, 256, 0, stream>>>(w_hh1, WHH1, G4 * HH);
  cvt_bf16<<<(G4 * HH + 255) / 256, 256, 0, stream>>>(w_ih2, WIH2, G4 * HH);
  cvt_bf16<<<(G4 * HH + 255) / 256, 256, 0, stream>>>(w_hh2, WHH2, G4 * HH);
  bias_fuse<<<(G4 + 255) / 256, 256, 0, stream>>>(b_ih1, b_hh1, B1, G4);
  bias_fuse<<<(G4 + 255) / 256, 256, 0, stream>>>(b_ih2, b_hh2, B2, G4);

  // --- zero all recurrent state ---
  hipMemsetAsync(ws + OFF_STATE, 0, STATE_BYTES, stream);

  // --- recurrence: 400 steps x 2 WMMA-fused SLSTM kernels ---
  const dim3 grid(HH / 32, BB / 32);  // (16, 4)
  for (int t = 0; t < TT; ++t) {
    const __bf16* xin = SPKC + (size_t)t * BB * OC;
    slstm_step<OC><<<grid, 256, 0, stream>>>(
        xin, M1BF[t & 1], WIH1, WHH1, B1, SYN1, MEM1,
        M1BF[(t + 1) & 1], SPK1, nullptr, thr1);
    slstm_step<HH><<<grid, 256, 0, stream>>>(
        SPK1, M2BF[t & 1], WIH2, WHH2, B2, SYN2, MEM2,
        M2BF[(t + 1) & 1], SPK2, FEAT, thr2);
  }

  // --- heads ---
  head_kernel<<<BB, 64, 0, stream>>>(FEAT, wg, bg, wd1, bd1, wd2, bd2, thrd,
                                     (float*)d_out);
}
